// Block_25366076850386
// MI455X (gfx1250) — matrix operand order repaired and checked by reading, not compiled, not run on hardware
//
#include <hip/hip_runtime.h>

// ---------------------------------------------------------------------------
// GPT block for MI455X (gfx1250, wave32). All matmuls via
// v_wmma_f32_16x16x32_bf16; GEMM weight tiles staged to LDS with the
// gfx1250 async global->LDS engine (ASYNCcnt), double-buffered.
// B=4, T=2048, C=1024, H=16, D=64, FF=4096.
// ---------------------------------------------------------------------------

typedef __attribute__((ext_vector_type(16))) __bf16 v16bf;
typedef __attribute__((ext_vector_type(8)))  float  v8f;

#define BS   4
#define TS   2048
#define CS   1024
#define HN   16
#define DH   64
#define FFS  4096
#define MS   (BS * TS)     // 8192 rows
#define C3S  (3 * CS)      // 3072

#if __has_builtin(__builtin_amdgcn_global_load_async_to_lds_b128) && \
    __has_builtin(__builtin_amdgcn_s_wait_asynccnt)
#define HAS_ASYNC_LDS 1
#else
#define HAS_ASYNC_LDS 0
#endif

union FragU { unsigned int u[8]; v16bf v; };

__device__ __forceinline__ unsigned short f2bf(float f) {
  unsigned int u = __float_as_uint(f);
  unsigned int r = u + 0x7fffu + ((u >> 16) & 1u);   // round-to-nearest-even
  if ((u & 0x7f800000u) == 0x7f800000u) r = u;       // inf/nan passthrough
  return (unsigned short)(r >> 16);
}

// 16 bytes global -> LDS (async DMA when available, else plain copy)
typedef int v4i_gl __attribute__((vector_size(16)));
typedef __attribute__((address_space(1))) v4i_gl* gptr128_t;
typedef __attribute__((address_space(3))) v4i_gl* lptr128_t;

__device__ __forceinline__ void stage_b128(const unsigned short* g,
                                           unsigned short* l) {
#if HAS_ASYNC_LDS
  __builtin_amdgcn_global_load_async_to_lds_b128(
      (gptr128_t)(g), (lptr128_t)(l), 0, 0);
#else
  *(uint4*)l = *(const uint4*)g;
#endif
}

// A-fragment: 16x32 bf16, M=lane%16, contiguous K: k = 16*(lane/16) + s.
__device__ __forceinline__ v16bf load_fragA(const unsigned short* __restrict__ A,
                                            size_t ld, int row, int k0, int lane) {
  const unsigned short* p = A + (size_t)row * ld;
  int kb = (lane < 16) ? 0 : 16;
  FragU f;
#pragma unroll
  for (int i = 0; i < 8; ++i) {
    int k = k0 + kb + 2 * i;
    f.u[i] = *(const unsigned int*)(p + k);           // K, K+1 packed
  }
  return f.v;
}

// B-fragment: 32x16 bf16 from W[N,K] row-major (global).
__device__ __forceinline__ v16bf load_fragB(const unsigned short* __restrict__ W,
                                            size_t ld, int col, int k0, int lane) {
  const unsigned short* p = W + (size_t)col * ld;
  int kb = (lane < 16) ? 0 : 16;
  FragU f;
#pragma unroll
  for (int j = 0; j < 8; ++j)
    f.u[j] = *(const unsigned int*)(p + k0 + kb + 2 * j);
  return f.v;
}

// B-fragment from an LDS tile laid out [col][k] (64 cols x 32 k, bf16).
__device__ __forceinline__ v16bf load_fragB_lds(const unsigned short* Bt,
                                                int col, int lane) {
  int kb = (lane < 16) ? 0 : 16;
  const unsigned short* p = Bt + col * 32 + kb;
  FragU f;
#pragma unroll
  for (int j = 0; j < 8; ++j)
    f.u[j] = *(const unsigned int*)(p + 2 * j);
  return f.v;
}

// ---------------------------------------------------------------------------
// fp32 -> bf16 elementwise
// ---------------------------------------------------------------------------
__global__ void k_f32_to_bf16(const float* __restrict__ in,
                              unsigned short* __restrict__ out, int n) {
  int i = blockIdx.x * blockDim.x + threadIdx.x;
  if (i < n) out[i] = f2bf(in[i]);
}

// ---------------------------------------------------------------------------
// LayerNorm over C, one block (256 thr) per row, bf16 output
// ---------------------------------------------------------------------------
__global__ void k_layernorm_bf16(const float* __restrict__ x,
                                 const float* __restrict__ w,
                                 const float* __restrict__ b,
                                 unsigned short* __restrict__ out) {
  int row = blockIdx.x;
  const float* xr = x + (size_t)row * CS;
  float s = 0.f, s2 = 0.f;
  for (int i = threadIdx.x; i < CS; i += blockDim.x) {
    float v = xr[i];
    s += v; s2 += v * v;
  }
#pragma unroll
  for (int m = 16; m >= 1; m >>= 1) {
    s  += __shfl_xor(s,  m, 32);
    s2 += __shfl_xor(s2, m, 32);
  }
  __shared__ float r0[8], r1[8];
  int wave = threadIdx.x >> 5, lane = threadIdx.x & 31;
  if (lane == 0) { r0[wave] = s; r1[wave] = s2; }
  __syncthreads();
  if (threadIdx.x == 0) {
    float t0 = 0.f, t1 = 0.f;
    for (int i = 0; i < 8; ++i) { t0 += r0[i]; t1 += r1[i]; }
    r0[0] = t0; r1[0] = t1;
  }
  __syncthreads();
  float mean = r0[0] * (1.0f / CS);
  float var  = r1[0] * (1.0f / CS) - mean * mean;
  float rstd = rsqrtf(var + 1e-5f);
  for (int i = threadIdx.x; i < CS; i += blockDim.x) {
    float v = (xr[i] - mean) * rstd * w[i] + b[i];
    out[(size_t)row * CS + i] = f2bf(v);
  }
}

// ---------------------------------------------------------------------------
// Generic WMMA GEMM: out = epilogue(A[M,K] * W[N,K]^T [+resid])
// Block 256 thr (8 waves): 256x64 tile; wave -> 32x64 strip
// (2 M-tiles x 4 N-tiles = 8 accumulators, 8 WMMAs per K-step of 32).
// The shared 64x32 B-tile is staged into LDS via async global->LDS DMA,
// double-buffered: DMA of step k+1 overlaps the WMMAs of step k.
// OUT_MODE 0: fp32 out = resid + acc
// OUT_MODE 1: bf16 out = gelu_exact(acc)
// OUT_MODE 2: bf16 out = acc
// ---------------------------------------------------------------------------
template <int OUT_MODE>
__global__ __launch_bounds__(256)
void k_gemm_wmma(const unsigned short* __restrict__ A,
                 const unsigned short* __restrict__ W,
                 const float* __restrict__ resid,
                 void* __restrict__ out,
                 int M, int N, int K) {
  int lane = threadIdx.x & 31;
  int wave = threadIdx.x >> 5;
  int m0 = blockIdx.x * 256 + wave * 32;
  int n0 = blockIdx.y * 64;
  int nl = lane & 15, half = lane >> 4;

  __shared__ unsigned short Blds[2][64 * 32];     // 2 x 4KB

  // cooperative staging: thread t -> 16B of W tile (col = t/4, kq = (t%4)*8)
  int scol = threadIdx.x >> 2;
  int skq  = (threadIdx.x & 3) * 8;
  const unsigned short* Wrow = W + (size_t)(n0 + scol) * K + skq;

  stage_b128(Wrow, &Blds[0][scol * 32 + skq]);    // prologue: tile for k0=0

  v8f acc[2][4] = {};
  int cur = 0;
  for (int k0 = 0; k0 < K; k0 += 32) {
    bool more = (k0 + 32 < K);
    if (more) {
      stage_b128(Wrow + k0 + 32, &Blds[cur ^ 1][scol * 32 + skq]);
#if HAS_ASYNC_LDS
      __builtin_amdgcn_s_wait_asynccnt(1);        // current tile complete
#endif
    } else {
#if HAS_ASYNC_LDS
      __builtin_amdgcn_s_wait_asynccnt(0);
#endif
    }
    __syncthreads();                              // publish all waves' quarters

    if (more) {                                   // prefetch next A lines
      __builtin_prefetch(A + (size_t)(m0 + nl) * K + k0 + 32, 0, 1);
      __builtin_prefetch(A + (size_t)(m0 + 16 + nl) * K + k0 + 32, 0, 1);
    }
    v16bf a0 = load_fragA(A, K, m0 + nl,      k0, lane);
    v16bf a1 = load_fragA(A, K, m0 + 16 + nl, k0, lane);
#pragma unroll
    for (int nt = 0; nt < 4; ++nt) {
      v16bf bf = load_fragB_lds(Blds[cur], nt * 16 + nl, lane);
      acc[0][nt] = __builtin_amdgcn_wmma_f32_16x16x32_bf16(
          false, a0, false, bf, (short)0, acc[0][nt], false, false);
      acc[1][nt] = __builtin_amdgcn_wmma_f32_16x16x32_bf16(
          false, a1, false, bf, (short)0, acc[1][nt], false, false);
    }
    __syncthreads();                              // tile free for restaging
    cur ^= 1;
  }

#pragma unroll
  for (int mt = 0; mt < 2; ++mt) {
#pragma unroll
    for (int nt = 0; nt < 4; ++nt) {
#pragma unroll
      for (int v = 0; v < 8; ++v) {
        int row = m0 + mt * 16 + v + 8 * half;
        int col = n0 + nt * 16 + nl;
        float val = acc[mt][nt][v];
        if (OUT_MODE == 0) {
          ((float*)out)[(size_t)row * N + col] =
              resid[(size_t)row * N + col] + val;
        } else if (OUT_MODE == 1) {
          float g = 0.5f * val * (1.0f + erff(val * 0.70710678118654752f));
          ((unsigned short*)out)[(size_t)row * N + col] = f2bf(g);
        } else {
          ((unsigned short*)out)[(size_t)row * N + col] = f2bf(val);
        }
      }
    }
  }
}

// ---------------------------------------------------------------------------
// Flash attention: one wave (block of 32) per 16 query rows per (b,h).
// qkv bf16 [B*T, 3C]; y bf16 [B*T, C]. Online softmax, 32 keys per step.
// ---------------------------------------------------------------------------
__global__ __launch_bounds__(32)
void k_attn_flash(const unsigned short* __restrict__ qkv,
                  unsigned short* __restrict__ y) {
  int lane = threadIdx.x;
  int nl = lane & 15, half = lane >> 4;
  int q0 = blockIdx.x * 16;
  int h  = blockIdx.y;
  int bb = blockIdx.z;

  size_t base = (size_t)bb * TS * C3S;
  const unsigned short* Qp = qkv + base + (size_t)h * DH;
  const unsigned short* Kp = qkv + base + CS + (size_t)h * DH;
  const unsigned short* Vp = qkv + base + 2 * CS + (size_t)h * DH;

  __shared__ unsigned short Plds[16 * 32];

  // Q fragments for d = 0..31 and 32..63
  int rowA = q0 + nl;
  v16bf qa0 = load_fragA(Qp, C3S, rowA, 0, lane);
  v16bf qa1 = load_fragA(Qp, C3S, rowA, 32, lane);

  v8f o[4] = {};
  float mrow[8], lrow[8];
#pragma unroll
  for (int v = 0; v < 8; ++v) { mrow[v] = -3.0e38f; lrow[v] = 0.f; }
  const float sc = 0.125f;   // 1/sqrt(64)

  for (int s0 = 0; s0 <= q0 + 15; s0 += 32) {
    // S = Q K^T for two 16-key tiles (each: 2 WMMAs over d)
    v8f sacc[2];
#pragma unroll
    for (int nt = 0; nt < 2; ++nt) {
      int colS = s0 + nt * 16 + nl;
      v16bf b0 = load_fragB(Kp, C3S, colS, 0, lane);
      v16bf b1 = load_fragB(Kp, C3S, colS, 32, lane);
      v8f z = {};
      z = __builtin_amdgcn_wmma_f32_16x16x32_bf16(false, qa0, false, b0,
                                                  (short)0, z, false, false);
      z = __builtin_amdgcn_wmma_f32_16x16x32_bf16(false, qa1, false, b1,
                                                  (short)0, z, false, false);
      sacc[nt] = z;
    }

    // online softmax per row (row = v + 8*half lives in one 16-lane group)
    float p0v[8], p1v[8], alpha[8];
#pragma unroll
    for (int v = 0; v < 8; ++v) {
      int r = q0 + v + 8 * half;
      float x0 = sacc[0][v] * sc; if (s0 + nl > r)      x0 = -3.0e38f;
      float x1 = sacc[1][v] * sc; if (s0 + 16 + nl > r) x1 = -3.0e38f;
      float mx = fmaxf(x0, x1);
#pragma unroll
      for (int m = 1; m < 16; m <<= 1) mx = fmaxf(mx, __shfl_xor(mx, m, 32));
      float nm = fmaxf(mrow[v], mx);
      float p0 = __expf(x0 - nm);
      float p1 = __expf(x1 - nm);
      float rs = p0 + p1;
#pragma unroll
      for (int m = 1; m < 16; m <<= 1) rs += __shfl_xor(rs, m, 32);
      alpha[v] = __expf(mrow[v] - nm);
      lrow[v]  = lrow[v] * alpha[v] + rs;
      mrow[v]  = nm;
      p0v[v] = p0; p1v[v] = p1;
    }

    // rescale running O by alpha (same (v,half)->row mapping as S)
#pragma unroll
    for (int nt = 0; nt < 4; ++nt)
#pragma unroll
      for (int v = 0; v < 8; ++v) o[nt][v] *= alpha[v];

    // transpose P (C/D layout) -> A layout through LDS, as bf16
    __syncthreads();
#pragma unroll
    for (int v = 0; v < 8; ++v) {
      int m = v + 8 * half;
      Plds[m * 32 + nl]      = f2bf(p0v[v]);
      Plds[m * 32 + 16 + nl] = f2bf(p1v[v]);
    }
    __syncthreads();
    v16bf pa;
    {
      FragU f;
      int kb = half ? 16 : 0;
#pragma unroll
      for (int i = 0; i < 8; ++i) {
        int k = kb + 2 * i;
        f.u[i] = *(const unsigned int*)(Plds + nl * 32 + k);
      }
      pa = f.v;
    }

    // O += P (16x32) * V (32x64): 4 WMMAs
#pragma unroll
    for (int nt = 0; nt < 4; ++nt) {
      FragU f;
      int kb = half ? 16 : 0;
      int d  = nt * 16 + nl;
#pragma unroll
      for (int j = 0; j < 8; ++j) {
        int s = s0 + kb + 2 * j;
        unsigned int lo = Vp[(size_t)s * C3S + d];
        unsigned int hi = Vp[(size_t)(s + 1) * C3S + d];
        f.u[j] = lo | (hi << 16);
      }
      o[nt] = __builtin_amdgcn_wmma_f32_16x16x32_bf16(
          false, pa, false, f.v, (short)0, o[nt], false, false);
    }
  }

  // normalize and store y[b*T + r, h*64 + d] as bf16
#pragma unroll
  for (int nt = 0; nt < 4; ++nt) {
#pragma unroll
    for (int v = 0; v < 8; ++v) {
      int r = q0 + v + 8 * half;
      int d = nt * 16 + nl;
      float val = o[nt][v] / lrow[v];
      y[(size_t)(bb * TS + r) * CS + h * DH + d] = f2bf(val);
    }
  }
}

// ---------------------------------------------------------------------------
// launch
// ---------------------------------------------------------------------------
extern "C" void kernel_launch(void* const* d_in, const int* in_sizes, int n_in,
                              void* d_out, int out_size, void* d_ws, size_t ws_size,
                              hipStream_t stream) {
  const float* x       = (const float*)d_in[0];
  const float* ln1_w   = (const float*)d_in[1];
  const float* ln1_b   = (const float*)d_in[2];
  const float* ln2_w   = (const float*)d_in[3];
  const float* ln2_b   = (const float*)d_in[4];
  const float* w_attn  = (const float*)d_in[5];
  const float* w_proj  = (const float*)d_in[6];
  const float* w_fc    = (const float*)d_in[7];
  const float* w_fcp   = (const float*)d_in[8];

  char* ws = (char*)d_ws;
  size_t off = 0;
  auto alloc = [&](size_t bytes) {
    char* p = ws + off;
    off = (off + bytes + 255) & ~(size_t)255;
    return p;
  };
  unsigned short* wa_bf  = (unsigned short*)alloc((size_t)C3S * CS * 2);
  unsigned short* wp_bf  = (unsigned short*)alloc((size_t)CS * CS * 2);
  unsigned short* wf_bf  = (unsigned short*)alloc((size_t)FFS * CS * 2);
  unsigned short* wfp_bf = (unsigned short*)alloc((size_t)CS * FFS * 2);
  unsigned short* bufA   = (unsigned short*)alloc((size_t)MS * CS * 2);   // h1 -> y -> h2
  unsigned short* qkv_bf = (unsigned short*)alloc((size_t)MS * C3S * 2);
  float*          x1     = (float*)alloc((size_t)MS * CS * 4);
  unsigned short* u_bf   = (unsigned short*)alloc((size_t)MS * FFS * 2);

  // 1) weights fp32 -> bf16
  auto cvt = [&](const float* src, unsigned short* dst, int n) {
    k_f32_to_bf16<<<(n + 255) / 256, 256, 0, stream>>>(src, dst, n);
  };
  cvt(w_attn, wa_bf,  C3S * CS);
  cvt(w_proj, wp_bf,  CS * CS);
  cvt(w_fc,   wf_bf,  FFS * CS);
  cvt(w_fcp,  wfp_bf, CS * FFS);

  // 2) h1 = LN1(x)
  k_layernorm_bf16<<<MS, 256, 0, stream>>>(x, ln1_w, ln1_b, bufA);

  // 3) qkv = h1 @ w_attn^T  [8192 x 3072]
  k_gemm_wmma<2><<<dim3(MS / 256, C3S / 64), 256, 0, stream>>>(
      bufA, wa_bf, nullptr, qkv_bf, MS, C3S, CS);

  // 4) y = attention(qkv)  (bf16, into bufA)
  k_attn_flash<<<dim3(TS / 16, HN, BS), 32, 0, stream>>>(qkv_bf, bufA);

  // 5) x1 = x + y @ w_proj^T
  k_gemm_wmma<0><<<dim3(MS / 256, CS / 64), 256, 0, stream>>>(
      bufA, wp_bf, x, x1, MS, CS, CS);

  // 6) h2 = LN2(x1)  (into bufA)
  k_layernorm_bf16<<<MS, 256, 0, stream>>>(x1, ln2_w, ln2_b, bufA);

  // 7) u = gelu(h2 @ w_fc^T)  [8192 x 4096]
  k_gemm_wmma<1><<<dim3(MS / 256, FFS / 64), 256, 0, stream>>>(
      bufA, wf_bf, nullptr, u_bf, MS, FFS, CS);

  // 8) out = x1 + u @ w_fc_proj^T
  k_gemm_wmma<0><<<dim3(MS / 256, CS / 64), 256, 0, stream>>>(
      u_bf, wfp_bf, x1, (float*)d_out, MS, CS, FFS);
}
